// ScaleDotProductAttention_16922171147122
// MI455X (gfx1250) — matrix operand order repaired and checked by reading, not compiled, not run on hardware
//
#include <hip/hip_runtime.h>

typedef __attribute__((ext_vector_type(16))) _Float16 v16h;
typedef __attribute__((ext_vector_type(8)))  float    v8f;
typedef __attribute__((ext_vector_type(4)))  float    v4f;

// Problem constants (B=16, S=4096, D=64 from the reference)
constexpr int kB       = 16;
constexpr int kS       = 4096;
constexpr int kD       = 64;
constexpr int kQTile   = 16;                 // query rows per workgroup
constexpr int kWaves   = 8;                  // wave32s per workgroup (2 per SIMD32)
constexpr int kThreads = kWaves * 32;        // 256
constexpr int kSRow    = kS + 4;             // LDS row pitch (pad 4 floats -> bank shift)
constexpr int kRedOff  = kQTile * kSRow;     // reduction scratch after score tile
constexpr int kRedSize = 16 * 16 + 16;       // [16][16] partials + [16] row values
constexpr int kPartOff = kRedOff + kRedSize; // phase-3 split-K partials: 4 tiles * 256
constexpr int kLdsFloats = kPartOff + 4 * 256;

__device__ __forceinline__ v8f wmma_f16(v16h a, v16h b, v8f c) {
    return __builtin_amdgcn_wmma_f32_16x16x32_f16(
        /*neg_a=*/false, a, /*neg_b=*/false, b,
        /*c_mod=*/(short)0, c, /*reuse_a=*/false, /*reuse_b=*/false);
}

// Convert 8 contiguous f32 (two float4 loads) into f16 fragment elems [base, base+8).
__device__ __forceinline__ void cvt8(v16h& dst, int base, const float* src) {
    const v4f lo = *reinterpret_cast<const v4f*>(src);
    const v4f hi = *reinterpret_cast<const v4f*>(src + 4);
    #pragma unroll
    for (int j = 0; j < 4; ++j) {
        dst[base + j]     = (_Float16)lo[j];
        dst[base + 4 + j] = (_Float16)hi[j];
    }
}

__global__ __launch_bounds__(kThreads)
void sdpa_wmma_kernel(const float* __restrict__ q,
                      const float* __restrict__ k,
                      const float* __restrict__ v,
                      float* __restrict__ ctx,     // [B,S,D]
                      float* __restrict__ attn)    // [B,S,S]
{
    extern __shared__ float lds[];
    float* scores = lds;                  // [16][kSRow]
    float* red    = lds + kRedOff;        // [16][16] partials + [16] row values
    float* part   = lds + kPartOff;       // [4][16][16] split-K partial C tiles

    const int tid  = threadIdx.x;
    const int lane = tid & 31;
    const int wave = tid >> 5;
    const int half = lane >> 4;           // 0 or 1 (lane group)
    const int l16  = lane & 15;

    const int bq = blockIdx.x;            // b*(S/16) + qtile
    const int b  = bq >> 8;               // S/16 == 256
    const int qt = bq & 255;
    const int q0 = qt * kQTile;

    const float* qb = q + ((size_t)b * kS + q0) * kD;
    const float* kb = k + (size_t)b * kS * kD;
    const float* vb = v + (size_t)b * kS * kD;

    // ---- Q A-fragments (16x32 A, same k order as the B fragments): per lane, K runs
    //      [half*16 .. +16) contiguous in elems 0..15 ----
    v16h aq0, aq1;
    {
        const float* qrow = qb + l16 * kD;       // A row M = lane%16
        cvt8(aq0, 0, qrow + half * 16);
        cvt8(aq0, 8, qrow + 8 + half * 16);
        cvt8(aq1, 0, qrow + 32 + half * 16);
        cvt8(aq1, 8, qrow + 40 + half * 16);
    }

    // ---- Phase 1: scores = (Q @ K^T) * 1/sqrt(D), 16x16 tile per WMMA pair ----
    const float scale = 0.125f;                  // 1/sqrt(64)
    for (int kb16 = wave; kb16 < kS / 16; kb16 += kWaves) {
        const float* krow = kb + (size_t)(kb16 * 16 + l16) * kD + half * 16;
        if (kb16 + kWaves < kS / 16)             // cover next tile's latency
            __builtin_prefetch(kb + (size_t)((kb16 + kWaves) * 16 + l16) * kD, 0, 0);
        // B fragment (32x16): lane holds key N=lane%16, 16 contiguous K values
        v16h b0, b1;
        cvt8(b0, 0, krow);      cvt8(b0, 8, krow + 8);     // K-dim 0..31
        cvt8(b1, 0, krow + 32); cvt8(b1, 8, krow + 40);    // K-dim 32..63
        v8f c = {};
        c = wmma_f16(aq0, b0, c);
        c = wmma_f16(aq1, b1, c);
        #pragma unroll
        for (int r = 0; r < 8; ++r) {
            const int row = r + half * 8;        // C/D layout: VGPR r -> M=r (+8 hi lanes)
            scores[row * kSRow + kb16 * 16 + l16] = c[r] * scale;
        }
    }
    __syncthreads();

    // ---- Phase 2: row softmax in LDS (16 threads per row, float4 sweeps) ----
    const int srow_id = tid >> 4;                // 0..15
    const int sub     = tid & 15;                // 0..15
    float* srow = scores + srow_id * kSRow;

    float m = -3.402823466e38f;
    for (int i = 0; i < kS / 64; ++i) {
        const v4f s4 = *reinterpret_cast<const v4f*>(srow + sub * 4 + 64 * i);
        m = fmaxf(fmaxf(fmaxf(m, s4[0]), fmaxf(s4[1], s4[2])), s4[3]);
    }
    red[srow_id * 16 + sub] = m;
    __syncthreads();
    if (sub == 0) {
        float mm = red[srow_id * 16];
        #pragma unroll
        for (int i = 1; i < 16; ++i) mm = fmaxf(mm, red[srow_id * 16 + i]);
        red[256 + srow_id] = mm;                 // row max
    }
    __syncthreads();
    const float rowmax = red[256 + srow_id];
    float ssum = 0.0f;
    for (int i = 0; i < kS / 64; ++i) {
        float* p = srow + sub * 4 + 64 * i;
        v4f s4 = *reinterpret_cast<const v4f*>(p);
        #pragma unroll
        for (int j = 0; j < 4; ++j) { s4[j] = __expf(s4[j] - rowmax); ssum += s4[j]; }
        *reinterpret_cast<v4f*>(p) = s4;
    }
    __syncthreads();                             // maxes fully consumed
    red[srow_id * 16 + sub] = ssum;
    __syncthreads();
    if (sub == 0) {
        float ss = red[srow_id * 16];
        #pragma unroll
        for (int i = 1; i < 16; ++i) ss += red[srow_id * 16 + i];
        red[256 + srow_id] = 1.0f / ss;          // row inverse-sum
    }
    __syncthreads();

    // ---- Normalize P in LDS and stream attention to HBM: 16B non-temporal
    //      stores (write-once 1 GB traffic must not evict L2-resident K/V) ----
    {
        float* arow = attn + ((size_t)b * kS + q0) * kS;
        for (int i4 = tid; i4 < kQTile * kS / 4; i4 += kThreads) {
            const int row = i4 >> 10;            // (i4*4)/4096
            const int col = (i4 << 2) & (kS - 1);
            const float inv = red[256 + row];
            float* p = scores + row * kSRow + col;
            v4f s4 = *reinterpret_cast<const v4f*>(p);
            #pragma unroll
            for (int j = 0; j < 4; ++j) s4[j] *= inv;
            *reinterpret_cast<v4f*>(p) = s4;     // normalized P for P@V
            __builtin_nontemporal_store(
                s4, reinterpret_cast<v4f*>(arow + (size_t)row * kS + col));
        }
    }
    __syncthreads();

    // ---- Phase 3: context = P @ V, split-K: wave = ksec*4 + ntile ----
    const int ntile = wave & 3;                  // output cols [ntile*16, +16)
    const int ksec  = wave >> 2;                 // K half: 0 -> [0,2048), 1 -> [2048,4096)
    const int ncol  = ntile * 16 + l16;
    v8f c2 = {};
    for (int kb32 = ksec * 64; kb32 < ksec * 64 + 64; ++kb32) {
        // A fragment from LDS P (row M = lane%16): two contiguous 8-float runs
        const float* prow = scores + l16 * kSRow + kb32 * 32;
        v16h a;
        cvt8(a, 0, prow + half * 16);
        cvt8(a, 8, prow + 8 + half * 16);
        // B fragment from V: column ncol, 16 consecutive K rows per lane half
        const float* vcol = vb + (size_t)(kb32 * 32 + half * 16) * kD + ncol;
        if (kb32 + 1 < ksec * 64 + 64)
            __builtin_prefetch(vb + (size_t)((kb32 + 1) * 32 + half * 16) * kD + ncol, 0, 0);
        v16h bv;
        #pragma unroll
        for (int j = 0; j < 16; ++j) bv[j] = (_Float16)vcol[(size_t)j * kD];
        c2 = wmma_f16(a, bv, c2);
    }
    if (ksec == 1) {                             // publish upper-K partials
        #pragma unroll
        for (int r = 0; r < 8; ++r)
            part[ntile * 256 + (r + half * 8) * 16 + l16] = c2[r];
    }
    __syncthreads();
    if (ksec == 0) {                             // combine and store context
        float* crow = ctx + ((size_t)b * kS + q0) * kD;
        #pragma unroll
        for (int r = 0; r < 8; ++r) {
            const int row = r + half * 8;
            crow[(size_t)row * kD + ncol] = c2[r] + part[ntile * 256 + row * 16 + l16];
        }
    }
}

extern "C" void kernel_launch(void* const* d_in, const int* in_sizes, int n_in,
                              void* d_out, int out_size, void* d_ws, size_t ws_size,
                              hipStream_t stream) {
    (void)in_sizes; (void)n_in; (void)d_ws; (void)ws_size; (void)out_size;
    const float* q = (const float*)d_in[0];
    const float* k = (const float*)d_in[1];
    const float* v = (const float*)d_in[2];
    float* ctx  = (float*)d_out;                                   // [B,S,D] first
    float* attn = (float*)d_out + (size_t)kB * kS * kD;            // then [B,S,S]

    const int grid = kB * (kS / kQTile);                           // 4096 blocks
    const size_t lds_bytes = (size_t)kLdsFloats * sizeof(float);   // ~268 KB (< 320 KB/WGP)
    sdpa_wmma_kernel<<<grid, kThreads, lds_bytes, stream>>>(q, k, v, ctx, attn);
}
